// PGCN_64845416235267
// MI455X (gfx1250) — compile-verified, matrix operand order checked by reading, not run on hardware
//
#include <hip/hip_runtime.h>
#include <hip/hip_bf16.h>
#include <stdint.h>

#define NN 50000
#define NE 600000
#define DD 128
#define NG 64

typedef __attribute__((ext_vector_type(16))) __bf16 v16bf;
typedef __attribute__((ext_vector_type(8)))  float v8f;

union Frag16 { v16bf v; unsigned int u[8]; uint4 q[2]; };

__device__ inline unsigned short f2bf(float x) {
    unsigned u = __float_as_uint(x);
    unsigned r = u + 0x7FFFu + ((u >> 16) & 1u);   // round-to-nearest-even
    return (unsigned short)(r >> 16);
}

__device__ inline void atomicMaxF(float* addr, float val) {
    if (val >= 0.0f) atomicMax((int*)addr, __float_as_int(val));
    else             atomicMin((unsigned int*)addr, __float_as_uint(val));
}

// ---------------------------------------------------------------- utilities
__global__ void fill_k(unsigned int* __restrict__ p, long long n, unsigned int v) {
    long long i = (long long)blockIdx.x * 256 + threadIdx.x;
    if (i < n) p[i] = v;
}

__global__ void cast_bf16_k(const float* __restrict__ in, unsigned short* __restrict__ out, long long n) {
    long long i = (long long)blockIdx.x * 256 + threadIdx.x;
    if (i < n) out[i] = f2bf(in[i]);
}

// ---------------------------------------------------------------- degrees
__global__ void deg1_k(const int* __restrict__ dst, const float* __restrict__ ea,
                       float* __restrict__ deg1, int nE) {
    int e = blockIdx.x * 256 + threadIdx.x;
    if (e >= nE) return;
    int d = dst[e];
#pragma unroll
    for (int c = 0; c < 4; ++c) atomicAdd(&deg1[c * NN + d], ea[e * 4 + c]);
}

__global__ void degu_k(const int* __restrict__ dst, float* __restrict__ degu, int nE) {
    int e = blockIdx.x * 256 + threadIdx.x;
    if (e >= nE) return;
    atomicAdd(&degu[dst[e]], 1.0f);
}

__global__ void rsqrt_k(float* __restrict__ p, int n) {
    int i = blockIdx.x * 256 + threadIdx.x;
    if (i < n) p[i] = rsqrtf(fmaxf(p[i], 1e-12f));
}

// ---------------------------------------------------------------- weight prep
__global__ void catW1_k(const float* __restrict__ A0, const float* __restrict__ A1,
                        const float* __restrict__ A2, const float* __restrict__ A3,
                        float* __restrict__ out) {
    int idx = blockIdx.x * 256 + threadIdx.x;           // 128*512
    if (idx >= 128 * 512) return;
    int k = idx >> 9, n = idx & 511, c = n >> 7, nn = n & 127;
    const float* W = (c == 0) ? A0 : (c == 1) ? A1 : (c == 2) ? A2 : A3;
    out[idx] = W[k * 128 + nn];
}

__global__ void catB1_k(const float* __restrict__ b0, const float* __restrict__ b1,
                        const float* __restrict__ b2, const float* __restrict__ b3,
                        float* __restrict__ out) {
    int f = blockIdx.x * 256 + threadIdx.x;
    if (f >= 512) return;
    int c = f >> 7, nn = f & 127;
    const float* b = (c == 0) ? b0 : (c == 1) ? b1 : (c == 2) ? b2 : b3;
    out[f] = b[nn];
}

// Pack W[K,N] (f32 row-major) into WMMA B fragments (bf16).
// Layout mirrors the ISA 16-bit A layout: lane = (half<<4)|(n%16),
// VGPR j holds K pair kk = (j<4) ? half*8+2j : 16+half*8+2(j-4).
// Stored so one lane's 8 dwords are contiguous (32B per lane, 1KB per fragment).
__global__ void packB_k(const float* __restrict__ W, unsigned int* __restrict__ Bp, int K, int N) {
    int idx = blockIdx.x * 256 + threadIdx.x;
    int total = (K >> 5) * (N >> 4) * 256;
    if (idx >= total) return;
    int j    = idx & 7;
    int lane = (idx >> 3) & 31;
    int t    = idx >> 8;
    int nT = N >> 4;
    int nt = t % nT, kb = t / nT;
    int n    = (nt << 4) + (lane & 15);
    int half = lane >> 4;
    int kk   = (j < 4) ? (half * 8 + 2 * j) : (16 + half * 8 + 2 * (j - 4));
    int k    = (kb << 5) + kk;
    unsigned lo = f2bf(W[(size_t)k * N + n]);
    unsigned hi = f2bf(W[(size_t)(k + 1) * N + n]);
    Bp[idx] = lo | (hi << 16);
}

// ---------------------------------------------------------------- WMMA GEMM
// C[M,N] = A_bf16[M,K] @ Bpacked[K,N], f32 accumulate.
// 8 waves / block; each wave computes a 16x64 strip (4 accumulators share 1 A frag).
__global__ __launch_bounds__(256) void gemm_bf16_wmma(
    const unsigned short* __restrict__ A, const uint4* __restrict__ Bp,
    float* __restrict__ C, int M, int N, int K)
{
    __shared__ uint4 lds[8][64];                 // 1KB A tile per wave
    const int lane = threadIdx.x & 31;
    const int wave = threadIdx.x >> 5;
    const int nT = N >> 4;
    const int groups = nT >> 2;                  // 4 N-tiles per wave
    const int t = blockIdx.x * 8 + wave;
    const int totalT = (M >> 4) * groups;
    if (t >= totalT) return;
    const int mt  = t / groups;
    const int nt0 = (t - mt * groups) << 2;
    const int m0  = mt << 4;
    const int row = lane & 15, half = lane >> 4;

    v8f acc[4];
#pragma unroll
    for (int c = 0; c < 4; ++c)
#pragma unroll
        for (int r = 0; r < 8; ++r) acc[c][r] = 0.0f;

    const int kSteps = K >> 5;
    for (int kb = 0; kb < kSteps; ++kb) {
        // stage 16x32 bf16 A tile in LDS (coalesced b128 loads)
#pragma unroll
        for (int j = 0; j < 2; ++j) {
            int idx = j * 32 + lane;             // uint4 index in tile (row = idx>>2)
            const uint4* gp = (const uint4*)(A + (size_t)(m0 + (idx >> 2)) * K + (size_t)kb * 32);
            lds[wave][idx] = gp[idx & 3];
        }
        if (kb + 1 < kSteps)
            __builtin_prefetch(A + (size_t)(m0 + row) * K + (size_t)(kb + 1) * 32, 0, 0);
        asm volatile("" ::: "memory");           // same-wave DS ops are in-order; stop reordering

        // A fragment per ISA 16-bit A layout: VGPR0-3 = K[half*8 .. +7], VGPR4-7 = K[16+half*8 .. +7]
        Frag16 a;
        a.q[0] = lds[wave][row * 4 + half];
        a.q[1] = lds[wave][row * 4 + 2 + half];

#pragma unroll
        for (int c = 0; c < 4; ++c) {
            Frag16 b;
            const uint4* bpp = Bp + (((size_t)kb * nT + nt0 + c) * 32 + lane) * 2;
            b.q[0] = bpp[0];
            b.q[1] = bpp[1];
            acc[c] = __builtin_amdgcn_wmma_f32_16x16x32_bf16(
                false, a.v, false, b.v, (short)0, acc[c], false, false);
        }
        asm volatile("" ::: "memory");
    }

    // C layout: VGPR r -> row = half*8 + r, col = lane%16
#pragma unroll
    for (int c = 0; c < 4; ++c) {
        float* cp = C + (size_t)(m0 + half * 8) * N + ((nt0 + c) << 4) + row;
#pragma unroll
        for (int r = 0; r < 8; ++r) cp[(size_t)r * N] = acc[c][r];
    }
}

// ---------------------------------------------------------------- edge scatter
__global__ __launch_bounds__(256) void scatter1_k(
    const float* __restrict__ h, const int* __restrict__ src, const int* __restrict__ dst,
    const float* __restrict__ ea, const float* __restrict__ dis1, float* __restrict__ agg)
{
    int e = blockIdx.x;
    int s = src[e], d = dst[e];
    int f = threadIdx.x;
#pragma unroll
    for (int it = 0; it < 2; ++it, f += 256) {
        int c = f >> 7;
        float norm = dis1[c * NN + s] * ea[e * 4 + c] * dis1[c * NN + d];
        atomicAdd(&agg[(size_t)d * 512 + f], h[(size_t)s * 512 + f] * norm);
    }
}

__global__ void scatterU_k(const float* __restrict__ h, const int* __restrict__ src,
                           const int* __restrict__ dst, const float* __restrict__ disu,
                           float* __restrict__ agg, int F)
{
    int e = blockIdx.x;
    int s = src[e], d = dst[e];
    float norm = disu[s] * disu[d];
    int f = threadIdx.x;
    atomicAdd(&agg[(size_t)d * F + f], h[(size_t)s * F + f] * norm);
}

// ---------------------------------------------------------------- node finalize
__global__ void finalize1_k(const float* __restrict__ agg, const float* __restrict__ h,
                            const float* __restrict__ dis1, const float* __restrict__ bias,
                            unsigned short* __restrict__ out)
{
    long long idx = (long long)blockIdx.x * 256 + threadIdx.x;
    if (idx >= (long long)NN * 512) return;
    int i = (int)(idx >> 9), f = (int)(idx & 511), c = f >> 7;
    float dv = dis1[c * NN + i];
    float v = agg[idx] + h[idx] * dv * dv + bias[f];
    out[idx] = f2bf(fmaxf(v, 0.0f));
}

__global__ void finalize2_k(const float* __restrict__ agg, const float* __restrict__ h,
                            const float* __restrict__ disu, const float* __restrict__ bias,
                            unsigned short* __restrict__ out)
{
    long long idx = (long long)blockIdx.x * 256 + threadIdx.x;
    if (idx >= (long long)NN * 256) return;
    int i = (int)(idx >> 8), f = (int)(idx & 255);
    float dv = disu[i];
    float v = agg[idx] + h[idx] * dv * dv + bias[f];
    out[idx] = f2bf(fmaxf(v, 0.0f));
}

__global__ void finalize3_k(float* __restrict__ agg, const float* __restrict__ h,
                            const float* __restrict__ disu, const float* __restrict__ bias)
{
    long long idx = (long long)blockIdx.x * 256 + threadIdx.x;
    if (idx >= (long long)NN * 128) return;
    int i = (int)(idx >> 7), f = (int)(idx & 127);
    float dv = disu[i];
    agg[idx] = agg[idx] + h[idx] * dv * dv + bias[f];   // f32, no relu
}

// ---------------------------------------------------------------- pooling + MLP
__global__ void pool_k(const float* __restrict__ h, const int* __restrict__ batch,
                       float* __restrict__ sums, float* __restrict__ maxs,
                       float* __restrict__ counts)
{
    int i = blockIdx.x;
    int f = threadIdx.x;                 // 128
    int b = batch[i];
    float v = h[(size_t)i * 128 + f];
    atomicAdd(&sums[b * 128 + f], v);
    atomicMaxF(&maxs[b * 128 + f], v);
    if (f == 0) atomicAdd(&counts[b], 1.0f);
}

__global__ void build_g_k(const float* __restrict__ sums, const float* __restrict__ maxs,
                          const float* __restrict__ counts, float* __restrict__ g)
{
    int idx = blockIdx.x * 256 + threadIdx.x;      // 64*256
    if (idx >= NG * 256) return;
    int gi = idx >> 8, f = idx & 255;
    float cnt = counts[gi];
    float v;
    if (f < 128) v = sums[gi * 128 + f] / fmaxf(cnt, 1.0f);
    else {
        v = maxs[gi * 128 + f - 128];
        v = (cnt > 0.0f) ? v : 0.0f;
    }
    g[idx] = v;
}

__global__ void mlp_k(const float* __restrict__ g, const float* __restrict__ Wm1,
                      const float* __restrict__ bm1, const float* __restrict__ Wm2,
                      const float* __restrict__ bm2, float* __restrict__ out)
{
    int gi = threadIdx.x;
    if (gi >= NG) return;
    float hid[8];
#pragma unroll
    for (int j = 0; j < 8; ++j) {
        float a = bm1[j];
        for (int k = 0; k < 256; ++k) a += g[gi * 256 + k] * Wm1[k * 8 + j];
        hid[j] = fmaxf(a, 0.0f);
    }
#pragma unroll
    for (int o = 0; o < 2; ++o) {
        float a = bm2[o];
#pragma unroll
        for (int j = 0; j < 8; ++j) a += hid[j] * Wm2[j * 2 + o];
        out[gi * 2 + o] = a;
    }
}

// ---------------------------------------------------------------- launch
extern "C" void kernel_launch(void* const* d_in, const int* in_sizes, int n_in,
                              void* d_out, int out_size, void* d_ws, size_t ws_size,
                              hipStream_t stream)
{
    const float* x   = (const float*)d_in[0];
    const int*   ei  = (const int*)d_in[1];
    const float* ea  = (const float*)d_in[2];
    const int*   bat = (const int*)d_in[3];
    const float *W1A = (const float*)d_in[4],  *b1A = (const float*)d_in[5];
    const float *W1B = (const float*)d_in[6],  *b1B = (const float*)d_in[7];
    const float *W1C = (const float*)d_in[8],  *b1C = (const float*)d_in[9];
    const float *W1D = (const float*)d_in[10], *b1D = (const float*)d_in[11];
    const float *W2  = (const float*)d_in[12], *b2  = (const float*)d_in[13];
    const float *W3  = (const float*)d_in[14], *b3  = (const float*)d_in[15];
    const float *Wm1 = (const float*)d_in[16], *bm1 = (const float*)d_in[17];
    const float *Wm2 = (const float*)d_in[18], *bm2 = (const float*)d_in[19];
    const int* src = ei;
    const int* dst = ei + NE;
    float* out = (float*)d_out;

    char* ws = (char*)d_ws;
    size_t off = 0;
    auto alloc = [&](size_t bytes) -> char* {
        char* p = ws + off;
        off += (bytes + 255) & ~(size_t)255;
        return p;
    };
    float*          h_buf  = (float*)alloc((size_t)NN * 512 * 4);  // h1 -> h2 -> h3
    float*          agg    = (float*)alloc((size_t)NN * 512 * 4);  // agg1/2/3, final h
    unsigned short* actbf  = (unsigned short*)alloc((size_t)NN * 512 * 2); // x/xcat/x2 (bf16)
    float*          w1cat  = (float*)alloc(128 * 512 * 4);
    float*          b1cat  = (float*)alloc(512 * 4);
    unsigned int*   bp1    = (unsigned int*)alloc((size_t)128 * 512 / 2 * 4);
    unsigned int*   bp2    = (unsigned int*)alloc((size_t)512 * 256 / 2 * 4);
    unsigned int*   bp3    = (unsigned int*)alloc((size_t)256 * 128 / 2 * 4);
    float*          dis    = (float*)alloc((size_t)5 * NN * 4);    // [4][NN] + [NN]
    float*          disu   = dis + 4 * NN;
    float*          sums   = (float*)alloc(NG * DD * 4);
    float*          maxs   = (float*)alloc(NG * DD * 4);
    float*          counts = (float*)alloc(NG * 4);
    float*          g      = (float*)alloc(NG * 256 * 4);

    // ---- degrees (init 1.0 = self-loop weight), then rsqrt in place
    fill_k<<<(5 * NN + 255) / 256, 256, 0, stream>>>((unsigned int*)dis, 5 * NN, 0x3F800000u);
    deg1_k<<<(NE + 255) / 256, 256, 0, stream>>>(dst, ea, dis, NE);
    degu_k<<<(NE + 255) / 256, 256, 0, stream>>>(dst, disu, NE);
    rsqrt_k<<<(5 * NN + 255) / 256, 256, 0, stream>>>(dis, 5 * NN);

    // ---- weight prep
    catW1_k<<<(128 * 512 + 255) / 256, 256, 0, stream>>>(W1A, W1B, W1C, W1D, w1cat);
    catB1_k<<<2, 256, 0, stream>>>(b1A, b1B, b1C, b1D, b1cat);
    packB_k<<<(128 * 512 / 2 + 255) / 256, 256, 0, stream>>>(w1cat, bp1, 128, 512);
    packB_k<<<(512 * 256 / 2 + 255) / 256, 256, 0, stream>>>(W2, bp2, 512, 256);
    packB_k<<<(256 * 128 / 2 + 255) / 256, 256, 0, stream>>>(W3, bp3, 256, 128);

    // ---- activations to bf16
    cast_bf16_k<<<(NN * DD + 255) / 256, 256, 0, stream>>>(x, actbf, (long long)NN * DD);

    // ---- layer 1 (4 fused convs): h1 = x @ [W1A|W1B|W1C|W1D]
    {
        int totalT = (NN / 16) * (512 / 64);
        gemm_bf16_wmma<<<(totalT + 7) / 8, 256, 0, stream>>>(actbf, (const uint4*)bp1, h_buf, NN, 512, 128);
    }
    fill_k<<<(int)(((long long)NN * 512 + 255) / 256), 256, 0, stream>>>((unsigned int*)agg, (long long)NN * 512, 0u);
    scatter1_k<<<NE, 256, 0, stream>>>(h_buf, src, dst, ea, dis, agg);
    finalize1_k<<<(int)(((long long)NN * 512 + 255) / 256), 256, 0, stream>>>(agg, h_buf, dis, b1cat, actbf);

    // ---- layer 2: h2 = xcat @ W2
    {
        int totalT = (NN / 16) * (256 / 64);
        gemm_bf16_wmma<<<(totalT + 7) / 8, 256, 0, stream>>>(actbf, (const uint4*)bp2, h_buf, NN, 256, 512);
    }
    fill_k<<<(int)(((long long)NN * 256 + 255) / 256), 256, 0, stream>>>((unsigned int*)agg, (long long)NN * 256, 0u);
    scatterU_k<<<NE, 256, 0, stream>>>(h_buf, src, dst, disu, agg, 256);
    finalize2_k<<<(int)(((long long)NN * 256 + 255) / 256), 256, 0, stream>>>(agg, h_buf, disu, b2, actbf);

    // ---- layer 3: h3 = x2 @ W3
    {
        int totalT = (NN / 16) * (128 / 64);
        gemm_bf16_wmma<<<(totalT + 7) / 8, 256, 0, stream>>>(actbf, (const uint4*)bp3, h_buf, NN, 128, 256);
    }
    fill_k<<<(int)(((long long)NN * 128 + 255) / 256), 256, 0, stream>>>((unsigned int*)agg, (long long)NN * 128, 0u);
    scatterU_k<<<NE, 128, 0, stream>>>(h_buf, src, dst, disu, agg, 128);
    finalize3_k<<<(int)(((long long)NN * 128 + 255) / 256), 256, 0, stream>>>(agg, h_buf, disu, b3);

    // ---- pooling + MLP head
    fill_k<<<(NG * DD + 255) / 256, 256, 0, stream>>>((unsigned int*)sums, NG * DD, 0u);
    fill_k<<<(NG * DD + 255) / 256, 256, 0, stream>>>((unsigned int*)maxs, NG * DD, 0xFF800000u);
    fill_k<<<1, 256, 0, stream>>>((unsigned int*)counts, NG, 0u);
    pool_k<<<NN, 128, 0, stream>>>(agg, bat, sums, maxs, counts);
    build_g_k<<<(NG * 256 + 255) / 256, 256, 0, stream>>>(sums, maxs, counts, g);
    mlp_k<<<1, 64, 0, stream>>>(g, Wm1, bm1, Wm2, bm2, out);
}
